// Net_49735721288249
// MI455X (gfx1250) — matrix operand order repaired and checked by reading, not compiled, not run on hardware
//
#include <hip/hip_runtime.h>
#include <math.h>

// ---------------------------------------------------------------------------
// CDNA5 (gfx1250) harmonic graph network.
// Core compute: per-edge complex contraction einsum('erc,roc->eo') lowered to
// V_WMMA_F32_16X16X32_BF16. 16 edges per wave (M), K = Mi*2*R*Ci with re/im as
// the fastest K axis; each 32-wide K step sits inside one (mp, ring) row, so
// the angular factor is a single LDS b64 load per step. Weights precomputed in
// WMMA-fragment-image order (32B/lane loads). f32 accumulate; segment-sum via
// global f32 atomics. Ring/chunk loops kept rolled to stay under 256 VGPRs.
// ---------------------------------------------------------------------------

typedef __attribute__((ext_vector_type(16))) __bf16 v16bf;
typedef __attribute__((ext_vector_type(8)))  float  v8f;

#define RINGS 6
#define MOUT  2            // MAX_ORDER + 1
#define EPSF  1e-12f

static __device__ inline v8f wmma_bf16(v16bf a, v16bf b, v8f c) {
  // v_wmma_f32_16x16x32_bf16  (neg_a, A, neg_b, B, c_mod, C, reuse_a, reuse_b)
  return __builtin_amdgcn_wmma_f32_16x16x32_bf16(false, a, false, b, (short)0, c,
                                                 false, false);
}

// ------------------------------- utility -----------------------------------

__global__ void zero_kernel(float* __restrict__ p, long long n) {
  long long i = (long long)blockIdx.x * blockDim.x + threadIdx.x;
  long long stride = (long long)gridDim.x * blockDim.x;
  for (; i < n; i += stride) p[i] = 0.0f;
}

// Build bf16 B-matrix in fragment-image order:
//   B[(((m*2+part)*NKS + ks)*NT + nt)*32 + lane][16]
// Fragment element `elem` of lane maps to k = ks*32 + (lane>>4)*16 + elem,
// o = nt*16 + (lane&15). Global K order: k = (((mp*RINGS + r)*Ci + c)*2 + reim).
// part 0 (yr): [Wr ; -Wi]   part 1 (yi): [Wi ; Wr].  o >= Co slots are zero.
__global__ void prep_w_kernel(const float* __restrict__ rad,
                              const float* __restrict__ ph,
                              __bf16* __restrict__ B,
                              int Mi, int Ci, int Co) {
  int NKS = (Mi * 2 * RINGS * Ci) / 32;
  int NT  = (Co + 15) / 16;
  int total = MOUT * 2 * NKS * NT * 512;
  int idx = blockIdx.x * blockDim.x + threadIdx.x;
  if (idx >= total) return;
  int elem = idx & 15;
  int lane = (idx >> 4) & 31;
  int nt   = (idx >> 9) % NT;
  int ks   = ((idx >> 9) / NT) % NKS;
  int part = (idx / (512 * NT * NKS)) & 1;
  int m    = idx / (512 * NT * NKS * 2);
  int o = nt * 16 + (lane & 15);
  int k = ks * 32 + (lane >> 4) * 16 + elem;
  int reim = k & 1;
  int kp = k >> 1;
  int c  = kp % Ci;
  int t  = kp / Ci;
  int r  = t % RINGS;
  int mp = t / RINGS;
  float val = 0.0f;
  if (o < Co) {
    int w = m * Mi + mp;
    float rv = rad[((size_t)(w * RINGS + r) * Co + o) * Ci + c];
    float wr, wi;
    if (ph) {
      float p = ph[((size_t)w * Co + o) * Ci + c];
      wr = rv * cosf(p);
      wi = rv * sinf(p);
    } else {
      wr = rv; wi = 0.0f;
    }
    val = (part == 0) ? ((reim == 0) ? wr : -wi)
                      : ((reim == 0) ? wi :  wr);
  }
  B[idx] = (__bf16)val;
}

// ------------------------- WMMA harmonic conv ------------------------------
// One wave per 16-edge tile. x: (N, MI, CI, 2). out: (N, MOUT, CO, 2),
// pre-zeroed. Launch guarantees E % 16 == 0 and CI % 16 == 0.
template<int MI, int CI, int CO>
__global__ __launch_bounds__(32) void hconv_kernel(
    const float* __restrict__ x, const int* __restrict__ eidx,
    const float* __restrict__ precomp, const float* __restrict__ conn,
    const v16bf* __restrict__ Bf, float* __restrict__ out, int E) {
  constexpr int NKS = (MI * 2 * RINGS * CI) / 32;
  constexpr int NT  = (CO + 15) / 16;
  constexpr int NCC = CI / 16;  // 16-pair channel chunks per (mp, ring) row
  __shared__ int    srcs[16];
  __shared__ float2 xs[16][MI][CI];        // gathered x[src] * conn^mp
  __shared__ float2 angs[16][RINGS][2];    // precomp (|k| in {0,1})

  const int lane = threadIdx.x;
  const int e0 = blockIdx.x * 16;

  if (lane < 16) srcs[lane] = eidx[E + e0 + lane];  // row 1 = src
  __syncthreads();

  // stage x_j (parallel-transport phase applied for mp==1)
  for (int i = lane; i < 16 * MI * CI; i += 32) {
    int el = i / (MI * CI);
    int rem = i % (MI * CI);
    int mp = rem / CI;
    int c = rem % CI;
    int e = e0 + el;
    const float* px = x + (((size_t)srcs[el] * MI + mp) * CI + c) * 2;
    float xr = px[0], xi = px[1];
    if (MI > 1 && mp == 1) {
      float cr = conn[2 * (size_t)e], cm = conn[2 * (size_t)e + 1];
      float nr = xr * cr - xi * cm;
      float ni = xr * cm + xi * cr;
      xr = nr; xi = ni;
    }
    xs[el][mp][c] = make_float2(xr, xi);
  }
  // stage angular precomp
  for (int i = lane; i < 16 * RINGS * 2; i += 32) {
    int el = i / (RINGS * 2);
    int rem = i % (RINGS * 2);
    int r = rem >> 1, kk = rem & 1;
    const float* pp = precomp + (((size_t)(e0 + el) * RINGS + r) * 2 + kk) * 2;
    angs[el][r][kk] = make_float2(pp[0], pp[1]);
  }
  __syncthreads();

  const int half = lane >> 4;   // K-split half for A/B layouts
  const int l15  = lane & 15;   // A row / B column

  // destination nodes for this lane's 8 accumulator rows (row = i + half*8)
  int node[8];
#pragma unroll
  for (int i = 0; i < 8; ++i) node[i] = eidx[e0 + i + half * 8];  // row 0 = dst

  const v8f zero8 = {0.f, 0.f, 0.f, 0.f, 0.f, 0.f, 0.f, 0.f};

  for (int m = 0; m < MOUT; ++m) {
    v8f accr[NT], acci[NT];
#pragma unroll
    for (int nt = 0; nt < NT; ++nt) { accr[nt] = zero8; acci[nt] = zero8; }

    const v16bf* Bm0 = Bf + (size_t)(m * 2    ) * NKS * NT * 32;
    const v16bf* Bm1 = Bf + (size_t)(m * 2 + 1) * NKS * NT * 32;

#pragma unroll
    for (int mp = 0; mp < MI; ++mp) {
      const int   absk = (m == mp) ? 0 : 1;
      const float sgn  = (m < mp) ? -1.0f : 1.0f;  // conj when m - mp < 0
#pragma unroll 1
      for (int r = 0; r < RINGS; ++r) {
        // one angular factor per 32-wide K step (single ds_load_b64)
        float2 av = angs[l15][r][absk];
        const float ar = av.x;
        const float ai = sgn * av.y;
#pragma unroll 1
        for (int cc = 0; cc < NCC; ++cc) {
          // ---- A fragment: pairs (2v,2v+1) share one complex input ----
          v16bf a;
#pragma unroll
          for (int v = 0; v < 8; ++v) {
            int c = cc * 16 + (half << 3) + v;
            float2 xv = xs[l15][mp][c];
            a[2 * v]     = (__bf16)(ar * xv.x - ai * xv.y);  // reim = 0
            a[2 * v + 1] = (__bf16)(ar * xv.y + ai * xv.x);  // reim = 1
          }
          // ---- B fragments (32B contiguous per lane) + WMMA ----
          int ks = (mp * RINGS + r) * NCC + cc;
#pragma unroll
          for (int nt = 0; nt < NT; ++nt) {
            v16bf br = Bm0[(ks * NT + nt) * 32 + lane];
            v16bf bi = Bm1[(ks * NT + nt) * 32 + lane];
            accr[nt] = wmma_bf16(a, br, accr[nt]);
            acci[nt] = wmma_bf16(a, bi, acci[nt]);
          }
        }
      }
    }

    // ---- scatter (segment-sum): C layout VGPR i -> M = i + half*8 ----
#pragma unroll
    for (int nt = 0; nt < NT; ++nt) {
      int o = nt * 16 + l15;
      bool ok = (CO % 16 == 0) || (o < CO);  // folds away for CO = 16/32
#pragma unroll
      for (int i = 0; i < 8; ++i) {
        if (ok) {
          size_t off = (((size_t)node[i] * MOUT + m) * CO + o) * 2;
          atomicAdd(out + off,     accr[nt][i]);
          atomicAdd(out + off + 1, acci[nt][i]);
        }
      }
    }
  }
}

// ---------------------------- elementwise ops ------------------------------

__global__ void lin0_kernel(const float* __restrict__ pos,
                            const float* __restrict__ w,
                            const float* __restrict__ b,
                            float* __restrict__ out, int N) {
  int idx = blockIdx.x * blockDim.x + threadIdx.x;
  if (idx >= N * 16) return;
  int o = idx % 16, n = idx / 16;
  float v = b[o];
  for (int k = 0; k < 3; ++k) v += pos[(size_t)n * 3 + k] * w[o * 3 + k];
  v = fmaxf(v, 0.0f);
  out[2 * (size_t)idx] = v;
  out[2 * (size_t)idx + 1] = 0.0f;
}

__global__ void nonlin_kernel(float* __restrict__ x, const float* __restrict__ b,
                              int total, int C) {
  int idx = blockIdx.x * blockDim.x + threadIdx.x;
  if (idx >= total) return;
  int c = idx % C;
  float xr = x[2 * (size_t)idx], xi = x[2 * (size_t)idx + 1];
  float mag = sqrtf(xr * xr + xi * xi + EPSF);
  float s = fmaxf(mag + b[c], 0.0f) / mag;
  x[2 * (size_t)idx]     = xr * s;
  x[2 * (size_t)idx + 1] = xi * s;
}

// out = nonlin(h2 + residual(x)); residual optionally projected, m-padded.
__global__ void resid_kernel(const float* __restrict__ h2,
                             const float* __restrict__ x,
                             const float* __restrict__ proj,
                             const float* __restrict__ nb,
                             float* __restrict__ out,
                             int N, int Cin, int Cout, int Min) {
  int idx = blockIdx.x * blockDim.x + threadIdx.x;
  int total = N * MOUT * Cout;
  if (idx >= total) return;
  int d = idx % Cout;
  int m = (idx / Cout) % MOUT;
  int n = idx / (Cout * MOUT);
  float hr = h2[2 * (size_t)idx], hi = h2[2 * (size_t)idx + 1];
  float rr = 0.0f, ri = 0.0f;
  if (m < Min) {
    if (proj) {
      for (int c = 0; c < Cin; ++c) {
        float w = proj[d * Cin + c];
        const float* px = x + (((size_t)n * Min + m) * Cin + c) * 2;
        rr += w * px[0];
        ri += w * px[1];
      }
    } else {
      const float* px = x + (((size_t)n * Min + m) * Cin + d) * 2;
      rr = px[0]; ri = px[1];
    }
  }
  float vr = hr + rr, vi = hi + ri;
  float mag = sqrtf(vr * vr + vi * vi + EPSF);
  float s = fmaxf(mag + nb[d], 0.0f) / mag;
  out[2 * (size_t)idx]     = vr * s;
  out[2 * (size_t)idx + 1] = vi * s;
}

// pool: transported scatter-mean fine(16ch) -> coarse(16ch)
__global__ void pool_scatter_kernel(const float* __restrict__ x,
                                    const int* __restrict__ parent,
                                    const float* __restrict__ pconn,
                                    float* __restrict__ sum, int N) {
  int idx = blockIdx.x * blockDim.x + threadIdx.x;
  if (idx >= N * 2 * 16) return;
  int c = idx % 16;
  int mp = (idx / 16) & 1;
  int n = idx / 32;
  float xr = x[2 * (size_t)idx], xi = x[2 * (size_t)idx + 1];
  if (mp == 1) {
    float pr = pconn[2 * (size_t)n], pi = pconn[2 * (size_t)n + 1];
    float nr = xr * pr - xi * pi;
    float ni = xr * pi + xi * pr;
    xr = nr; xi = ni;
  }
  int p = parent[n];
  size_t off = (((size_t)p * 2 + mp) * 16 + c) * 2;
  atomicAdd(sum + off,     xr);
  atomicAdd(sum + off + 1, xi);
}

__global__ void pool_count_kernel(const int* __restrict__ parent,
                                  float* __restrict__ cnt, int N) {
  int n = blockIdx.x * blockDim.x + threadIdx.x;
  if (n < N) atomicAdd(&cnt[parent[n]], 1.0f);
}

__global__ void pool_div_kernel(float* __restrict__ sum,
                                const float* __restrict__ cnt, int Nc) {
  int idx = blockIdx.x * blockDim.x + threadIdx.x;
  if (idx >= Nc * 2 * 16) return;
  int p = idx / 32;
  float d = fmaxf(cnt[p], 1.0f);
  sum[2 * (size_t)idx]     /= d;
  sum[2 * (size_t)idx + 1] /= d;
}

// unpool coarse(32ch) into first 32 channels of the 48ch concat buffer
__global__ void unpool_kernel(const float* __restrict__ xc,
                              const int* __restrict__ parent,
                              const float* __restrict__ pconn,
                              float* __restrict__ xcat, int N) {
  int idx = blockIdx.x * blockDim.x + threadIdx.x;
  if (idx >= N * 2 * 32) return;
  int c = idx % 32;
  int mp = (idx / 32) & 1;
  int n = idx / 64;
  int p = parent[n];
  const float* px = xc + (((size_t)p * 2 + mp) * 32 + c) * 2;
  float vr = px[0], vi = px[1];
  if (mp == 1) {  // conjugated transport back down
    float pr = pconn[2 * (size_t)n], pi = -pconn[2 * (size_t)n + 1];
    float nr = vr * pr - vi * pi;
    float ni = vr * pi + vi * pr;
    vr = nr; vi = ni;
  }
  size_t off = (((size_t)n * 2 + mp) * 48 + c) * 2;
  xcat[off] = vr; xcat[off + 1] = vi;
}

// copy x_pre(16ch) into channels [32,48) of concat buffer
__global__ void copy_pre_kernel(const float* __restrict__ xp,
                                float* __restrict__ xcat, int N) {
  int idx = blockIdx.x * blockDim.x + threadIdx.x;
  if (idx >= N * 2 * 16) return;
  int c = idx % 16;
  int mp = (idx / 16) & 1;
  int n = idx / 32;
  size_t off = (((size_t)n * 2 + mp) * 48 + 32 + c) * 2;
  xcat[off]     = xp[2 * (size_t)idx];
  xcat[off + 1] = xp[2 * (size_t)idx + 1];
}

// magnitude head + log_softmax over 8 classes
__global__ void head_kernel(const float* __restrict__ xf,
                            const float* __restrict__ bias,
                            float* __restrict__ out, int N) {
  int n = blockIdx.x * blockDim.x + threadIdx.x;
  if (n >= N) return;
  float l[8];
  float mx = -1e30f;
  for (int o = 0; o < 8; ++o) {
    float s = bias[o];
    for (int m = 0; m < 2; ++m) {
      const float* p = xf + (((size_t)n * 2 + m) * 8 + o) * 2;
      s += sqrtf(p[0] * p[0] + p[1] * p[1] + EPSF);
    }
    l[o] = s;
    mx = fmaxf(mx, s);
  }
  float se = 0.0f;
  for (int o = 0; o < 8; ++o) se += expf(l[o] - mx);
  float lse = logf(se);
  for (int o = 0; o < 8; ++o) out[(size_t)n * 8 + o] = l[o] - mx - lse;
}

// ------------------------------ host side ----------------------------------

#define N0 10000
#define E0 160000
#define N1 2500
#define E1 40000

static inline int ceil_div(long long a, int b) { return (int)((a + b - 1) / b); }

static void zero_f(float* p, long long n, hipStream_t s) {
  int blocks = ceil_div(n, 256);
  if (blocks > 65535) blocks = 65535;
  zero_kernel<<<blocks, 256, 0, s>>>(p, n);
}

static void prep_w(const float* rad, const float* ph, __bf16* B,
                   int Mi, int Ci, int Co, hipStream_t s) {
  int NKS = (Mi * 2 * RINGS * Ci) / 32;
  int NT  = (Co + 15) / 16;
  int total = MOUT * 2 * NKS * NT * 512;
  prep_w_kernel<<<ceil_div(total, 256), 256, 0, s>>>(rad, ph, B, Mi, Ci, Co);
}

static void nonlin(float* x, const float* b, int N, int M, int C, hipStream_t s) {
  int total = N * M * C;
  nonlin_kernel<<<ceil_div(total, 256), 256, 0, s>>>(x, b, total, C);
}

extern "C" void kernel_launch(void* const* d_in, const int* in_sizes, int n_in,
                              void* d_out, int out_size, void* d_ws, size_t ws_size,
                              hipStream_t stream) {
  (void)in_sizes; (void)n_in; (void)out_size; (void)ws_size;
  // ---- inputs (setup_inputs insertion order; params flattened in
  //      _make_params insertion order: rad1, ph1, rad2, ph2, nb1, nb2[, proj]) ----
  const float* pos   = (const float*)d_in[0];
  const int*   ei0   = (const int*)  d_in[1];
  const float* pre0  = (const float*)d_in[2];
  const float* conn0 = (const float*)d_in[3];
  const int*   ei1   = (const int*)  d_in[4];
  const float* pre1  = (const float*)d_in[5];
  const float* conn1 = (const float*)d_in[6];
  const int*   ppar  = (const int*)  d_in[7];
  const float* pconn = (const float*)d_in[8];
  const float* lin0w = (const float*)d_in[9];
  const float* lin0b = (const float*)d_in[10];
  // blocks: b11@11, b12@17, b21@23(+proj), b22@30, b31@36, b32@42,
  //         b41@48(+proj), b42@55, final_rad@61, bias@62
  const float* B11[6]; for (int i = 0; i < 6; ++i) B11[i] = (const float*)d_in[11 + i];
  const float* B12[6]; for (int i = 0; i < 6; ++i) B12[i] = (const float*)d_in[17 + i];
  const float* B21[7]; for (int i = 0; i < 7; ++i) B21[i] = (const float*)d_in[23 + i];
  const float* B22[6]; for (int i = 0; i < 6; ++i) B22[i] = (const float*)d_in[30 + i];
  const float* B31[6]; for (int i = 0; i < 6; ++i) B31[i] = (const float*)d_in[36 + i];
  const float* B32[6]; for (int i = 0; i < 6; ++i) B32[i] = (const float*)d_in[42 + i];
  const float* B41[7]; for (int i = 0; i < 7; ++i) B41[i] = (const float*)d_in[48 + i];
  const float* B42[6]; for (int i = 0; i < 6; ++i) B42[i] = (const float*)d_in[55 + i];
  const float* frad = (const float*)d_in[61];
  const float* bias = (const float*)d_in[62];
  float* out = (float*)d_out;

  // ---- workspace layout (floats) ----
  float* base = (float*)d_ws;
  float* F0   = base + 0;        // 640000  (N0,2,16,2)
  float* F1   = base + 640000;
  float* F2   = base + 1280000;  // x_pre lives here
  float* F3   = base + 1920000;
  float* XCAT = base + 2560000;  // 1,920,000 (N0,2,48,2)
  float* CO0  = base + 4480000;  // 320000 (N1,2,32,2)
  float* CO1  = base + 4800000;
  float* CO2  = base + 5120000;
  float* CNT  = base + 5440000;  // 2500
  float* FIN  = base + 5443000;  // 320000 (N0,2,8,2)
  __bf16* WBraw = (__bf16*)(base + 5763000);  // bf16 fragment staging (32B aligned)
  const v16bf* WBf = (const v16bf*)WBraw;

  const long long F16SZ  = (long long)N0 * 2 * 16 * 2;
  const long long CO32SZ = (long long)N1 * 2 * 32 * 2;

  // ---- lin0: x = relu(pos @ W^T + b) as complex (N0,1,16,2) ----
  lin0_kernel<<<ceil_div(N0 * 16, 256), 256, 0, stream>>>(pos, lin0w, lin0b, F0, N0);

  dim3 g0(E0 / 16), g1(E1 / 16), wv(32);

  // ---- b11 (16->16, Mi=1 then 2) ----
  prep_w(B11[0], B11[1], WBraw, 1, 16, 16, stream);
  zero_f(F1, F16SZ, stream);
  hconv_kernel<1, 16, 16><<<g0, wv, 0, stream>>>(F0, ei0, pre0, conn0, WBf, F1, E0);
  nonlin(F1, B11[4], N0, 2, 16, stream);
  prep_w(B11[2], B11[3], WBraw, 2, 16, 16, stream);
  zero_f(F3, F16SZ, stream);
  hconv_kernel<2, 16, 16><<<g0, wv, 0, stream>>>(F1, ei0, pre0, conn0, WBf, F3, E0);
  resid_kernel<<<ceil_div(N0 * 2 * 16, 256), 256, 0, stream>>>(
      F3, F0, nullptr, B11[5], F1, N0, 16, 16, 1);  // x -> F1

  // ---- b12 (16->16) ----
  prep_w(B12[0], B12[1], WBraw, 2, 16, 16, stream);
  zero_f(F3, F16SZ, stream);
  hconv_kernel<2, 16, 16><<<g0, wv, 0, stream>>>(F1, ei0, pre0, conn0, WBf, F3, E0);
  nonlin(F3, B12[4], N0, 2, 16, stream);
  prep_w(B12[2], B12[3], WBraw, 2, 16, 16, stream);
  zero_f(F0, F16SZ, stream);
  hconv_kernel<2, 16, 16><<<g0, wv, 0, stream>>>(F3, ei0, pre0, conn0, WBf, F0, E0);
  resid_kernel<<<ceil_div(N0 * 2 * 16, 256), 256, 0, stream>>>(
      F0, F1, nullptr, B12[5], F2, N0, 16, 16, 2);  // x_pre -> F2

  // ---- pool: fine (N0,2,16,2) -> coarse (N1,2,16,2) in CO0 ----
  zero_f(CO0, (long long)N1 * 2 * 16 * 2, stream);
  zero_f(CNT, N1, stream);
  pool_scatter_kernel<<<ceil_div(N0 * 2 * 16, 256), 256, 0, stream>>>(F2, ppar, pconn, CO0, N0);
  pool_count_kernel<<<ceil_div(N0, 256), 256, 0, stream>>>(ppar, CNT, N0);
  pool_div_kernel<<<ceil_div(N1 * 2 * 16, 256), 256, 0, stream>>>(CO0, CNT, N1);

  // ---- b21 (16->32, proj) ----
  prep_w(B21[0], B21[1], WBraw, 2, 16, 32, stream);
  zero_f(CO1, CO32SZ, stream);
  hconv_kernel<2, 16, 32><<<g1, wv, 0, stream>>>(CO0, ei1, pre1, conn1, WBf, CO1, E1);
  nonlin(CO1, B21[4], N1, 2, 32, stream);
  prep_w(B21[2], B21[3], WBraw, 2, 32, 32, stream);
  zero_f(CO2, CO32SZ, stream);
  hconv_kernel<2, 32, 32><<<g1, wv, 0, stream>>>(CO1, ei1, pre1, conn1, WBf, CO2, E1);
  resid_kernel<<<ceil_div(N1 * 2 * 32, 256), 256, 0, stream>>>(
      CO2, CO0, B21[6], B21[5], CO1, N1, 16, 32, 2);  // xc -> CO1

  // ---- b22 / b31 / b32 (32->32) ----
  const float* const* blks[3] = {B22, B31, B32};
  float* xc = CO1;
  float* spare = CO0;
  for (int bi = 0; bi < 3; ++bi) {
    const float* const* P = blks[bi];
    prep_w(P[0], P[1], WBraw, 2, 32, 32, stream);
    zero_f(spare, CO32SZ, stream);
    hconv_kernel<2, 32, 32><<<g1, wv, 0, stream>>>(xc, ei1, pre1, conn1, WBf, spare, E1);
    nonlin(spare, P[4], N1, 2, 32, stream);
    prep_w(P[2], P[3], WBraw, 2, 32, 32, stream);
    zero_f(CO2, CO32SZ, stream);
    hconv_kernel<2, 32, 32><<<g1, wv, 0, stream>>>(spare, ei1, pre1, conn1, WBf, CO2, E1);
    resid_kernel<<<ceil_div(N1 * 2 * 32, 256), 256, 0, stream>>>(
        CO2, xc, nullptr, P[5], spare, N1, 32, 32, 2);
    float* t = xc; xc = spare; spare = t;  // new xc in `spare`, old xc freed
  }
  // xc now holds the b32 output

  // ---- unpool + concat into XCAT (N0,2,48,2): [xf(32) | x_pre(16)] ----
  unpool_kernel<<<ceil_div(N0 * 2 * 32, 256), 256, 0, stream>>>(xc, ppar, pconn, XCAT, N0);
  copy_pre_kernel<<<ceil_div(N0 * 2 * 16, 256), 256, 0, stream>>>(F2, XCAT, N0);

  // ---- b41 (48->16, proj) ----
  prep_w(B41[0], B41[1], WBraw, 2, 48, 16, stream);
  zero_f(F0, F16SZ, stream);
  hconv_kernel<2, 48, 16><<<g0, wv, 0, stream>>>(XCAT, ei0, pre0, conn0, WBf, F0, E0);
  nonlin(F0, B41[4], N0, 2, 16, stream);
  prep_w(B41[2], B41[3], WBraw, 2, 16, 16, stream);
  zero_f(F1, F16SZ, stream);
  hconv_kernel<2, 16, 16><<<g0, wv, 0, stream>>>(F0, ei0, pre0, conn0, WBf, F1, E0);
  resid_kernel<<<ceil_div(N0 * 2 * 16, 256), 256, 0, stream>>>(
      F1, XCAT, B41[6], B41[5], F3, N0, 48, 16, 2);  // x -> F3

  // ---- b42 (16->16) ----
  prep_w(B42[0], B42[1], WBraw, 2, 16, 16, stream);
  zero_f(F0, F16SZ, stream);
  hconv_kernel<2, 16, 16><<<g0, wv, 0, stream>>>(F3, ei0, pre0, conn0, WBf, F0, E0);
  nonlin(F0, B42[4], N0, 2, 16, stream);
  prep_w(B42[2], B42[3], WBraw, 2, 16, 16, stream);
  zero_f(F1, F16SZ, stream);
  hconv_kernel<2, 16, 16><<<g0, wv, 0, stream>>>(F0, ei0, pre0, conn0, WBf, F1, E0);
  resid_kernel<<<ceil_div(N0 * 2 * 16, 256), 256, 0, stream>>>(
      F1, F3, nullptr, B42[5], F0, N0, 16, 16, 2);  // x -> F0

  // ---- final conv (16->8, real weights) + head ----
  prep_w(frad, nullptr, WBraw, 2, 16, 8, stream);
  zero_f(FIN, (long long)N0 * 2 * 8 * 2, stream);
  hconv_kernel<2, 16, 8><<<g0, wv, 0, stream>>>(F0, ei0, pre0, conn0, WBf, FIN, E0);
  head_kernel<<<ceil_div(N0, 256), 256, 0, stream>>>(FIN, bias, out, N0);
}
